// CoreAttention_31327491457438
// MI455X (gfx1250) — matrix operand order repaired and checked by reading, not compiled, not run on hardware
//
#include <hip/hip_runtime.h>

typedef __attribute__((ext_vector_type(16))) _Float16 v16h;
typedef __attribute__((ext_vector_type(8)))  _Float16 v8h;
typedef __attribute__((ext_vector_type(8)))  float    v8f;

#define B_    2
#define S_    2048
#define H_    16
#define D_    64
#define SCALE 8.0f      // sqrt(D) -- reference MULTIPLIES q by sqrt(d_head)
#define CHUNK 32        // kv rows per iteration
#define WAVES 4

// ---- DPP16 reductions within each 16-lane half (a DPP "row" == our matrix row span).
// quad_perm(1,0,3,2)=0xB1, quad_perm(2,3,0,1)=0x4E, row_half_mirror=0x141, row_mirror=0x140.
// All four are full permutations of the 16-lane row: pure VALU, no DS traffic.
__device__ __forceinline__ float red_max16(float x) {
    int t;
    t = __builtin_amdgcn_update_dpp(__float_as_int(x), __float_as_int(x), 0xB1, 0xF, 0xF, true);
    x = fmaxf(x, __int_as_float(t));
    t = __builtin_amdgcn_update_dpp(__float_as_int(x), __float_as_int(x), 0x4E, 0xF, 0xF, true);
    x = fmaxf(x, __int_as_float(t));
    t = __builtin_amdgcn_update_dpp(__float_as_int(x), __float_as_int(x), 0x141, 0xF, 0xF, true);
    x = fmaxf(x, __int_as_float(t));
    t = __builtin_amdgcn_update_dpp(__float_as_int(x), __float_as_int(x), 0x140, 0xF, 0xF, true);
    x = fmaxf(x, __int_as_float(t));
    return x;
}
__device__ __forceinline__ float red_sum16(float x) {
    int t;
    t = __builtin_amdgcn_update_dpp(__float_as_int(x), __float_as_int(x), 0xB1, 0xF, 0xF, true);
    x += __int_as_float(t);
    t = __builtin_amdgcn_update_dpp(__float_as_int(x), __float_as_int(x), 0x4E, 0xF, 0xF, true);
    x += __int_as_float(t);
    t = __builtin_amdgcn_update_dpp(__float_as_int(x), __float_as_int(x), 0x141, 0xF, 0xF, true);
    x += __int_as_float(t);
    t = __builtin_amdgcn_update_dpp(__float_as_int(x), __float_as_int(x), 0x140, 0xF, 0xF, true);
    x += __int_as_float(t);
    return x;
}

// convert 16 consecutive f32 (LDS) -> f16 fragment
__device__ __forceinline__ v16h cvt16(const float* p) {
    v16h r;
    #pragma unroll
    for (int e = 0; e < 16; ++e) r[e] = (_Float16)p[e];
    return r;
}

__global__ __launch_bounds__(WAVES * 32)
void core_attn_fwd(const float* __restrict__ q,
                   const float* __restrict__ k,
                   const float* __restrict__ v,
                   const float* __restrict__ bias,
                   const unsigned char* __restrict__ mask,
                   float* __restrict__ out)
{
    // K chunk [t][d] f32 (8KB, filled by async DMA), V chunk transposed [d][t] f16 (4KB),
    // per-wave P scratch 16x32 f16 (4KB)
    __shared__ __align__(64) float     kF[CHUNK * D_];
    __shared__ __align__(32) _Float16  vT[D_ * CHUNK];
    __shared__ __align__(32) _Float16  pT[WAVES][16 * CHUNK];

    const int tid    = threadIdx.x;
    const int wave   = tid >> 5;
    const int lane   = tid & 31;
    const int l15    = lane & 15;
    const int hi     = (lane >> 4) & 1;   // lane half: 0 = lanes 0-15, 1 = 16-31
    const int half8  = hi * 8;
    const int half16 = hi * 16;

    const int b  = blockIdx.z;
    const int h  = blockIdx.y;
    const int q0 = blockIdx.x * (WAVES * 16) + wave * 16;

    // ---- Q fragment (A-matrix layout, 16x32 f16 per d-half), scale folded in ----
    // A layout: lane L holds row (L&15); elem e -> K = 16*half + e
    v16h qa0, qa1;
    {
        const float* qb = q + (((size_t)b * S_ + (q0 + l15)) * H_ + h) * D_;
        #pragma unroll
        for (int e = 0; e < 16; ++e) {
            const int dk = half16 + e;
            qa0[e] = (_Float16)(qb[dk]      * SCALE);
            qa1[e] = (_Float16)(qb[32 + dk] * SCALE);
        }
    }

    // per-lane streaming base pointers (row = q0 + half8, col = l15); j steps by
    // immediate offsets j*S_, chunk steps by +CHUNK
    const float*         bp = bias + ((((size_t)b * H_ + h) * S_) + q0 + half8) * S_ + l15;
    const unsigned char* mp = mask + (((size_t)b * S_) + q0 + half8) * S_ + l15;

    float m[8], l[8];
    v8f acc0 = {}, acc1 = {}, acc2 = {}, acc3 = {};
    #pragma unroll
    for (int j = 0; j < 8; ++j) { m[j] = -1e30f; l[j] = 0.0f; }

    const int srow = tid >> 2;          // 0..31: kv row this thread stages
    const int dseg = (tid & 3) * 16;    // 16-float segment of d

    // LDS byte offset of this thread's K staging slot (generic->LDS offset truncation)
    const unsigned kLds = (unsigned)(size_t)&kF[srow * D_ + dseg];

    for (int c = 0; c < S_ / CHUNK; ++c) {
        const int t0 = c * CHUNK;

        __syncthreads();   // all readers of previous K/V tiles are done
        {
            const size_t gro = (((size_t)b * S_ + t0 + srow) * H_ + h) * D_ + dseg;

            // K: async DMA 64B straight into LDS (f32, converted at fragment build)
            const unsigned long long ga = (unsigned long long)(size_t)(k + gro);
            asm volatile(
                "global_load_async_to_lds_b128 %0, %1, off\n\t"
                "global_load_async_to_lds_b128 %0, %1, off offset:16\n\t"
                "global_load_async_to_lds_b128 %0, %1, off offset:32\n\t"
                "global_load_async_to_lds_b128 %0, %1, off offset:48"
                :: "v"(kLds), "v"(ga) : "memory");

            // V: explicit load-convert-transpose (DMA cannot transpose)
            const float* vp = v + gro;
            #pragma unroll
            for (int i = 0; i < 16; ++i)
                vT[(dseg + i) * CHUNK + srow] = (_Float16)vp[i];

            asm volatile("s_wait_asynccnt 0" ::: "memory");
        }
        __syncthreads();

        if (c + 1 < S_ / CHUNK) {   // prefetch next bias chunk (dominant stream)
            __builtin_prefetch(bp + t0 + CHUNK, 0, 1);
        }

        // ---- S = (scale*Q) . K^T : B layout: lane L = col (L&15); elem e -> K = 16*half + e
        const float* kr0 = &kF[l15 * D_];
        const float* kr1 = &kF[(16 + l15) * D_];
        v16h kb00 = cvt16(kr0 + half16);        // t-cols 0-15,  d 0-31
        v16h kb01 = cvt16(kr0 + 32 + half16);   // t-cols 0-15,  d 32-63
        v16h kb10 = cvt16(kr1 + half16);        // t-cols 16-31, d 0-31
        v16h kb11 = cvt16(kr1 + 32 + half16);

        v8f s0 = {}, s1 = {};
        s0 = __builtin_amdgcn_wmma_f32_16x16x32_f16(false, qa0, false, kb00, (short)0, s0, false, false);
        s0 = __builtin_amdgcn_wmma_f32_16x16x32_f16(false, qa1, false, kb01, (short)0, s0, false, false);
        s1 = __builtin_amdgcn_wmma_f32_16x16x32_f16(false, qa0, false, kb10, (short)0, s1, false, false);
        s1 = __builtin_amdgcn_wmma_f32_16x16x32_f16(false, qa1, false, kb11, (short)0, s1, false, false);

        // ---- bias + masked_fill(0) + online softmax (C layout: row = j + 8*half) ----
        #pragma unroll
        for (int j = 0; j < 8; ++j) {
            const float         x0b = bp[(size_t)j * S_ + t0];
            const float         x1b = bp[(size_t)j * S_ + t0 + 16];
            const unsigned char m0  = mp[(size_t)j * S_ + t0];
            const unsigned char m1  = mp[(size_t)j * S_ + t0 + 16];
            const float x0 = m0 ? 0.0f : s0[j] + x0b;
            const float x1 = m1 ? 0.0f : s1[j] + x1b;

            const float mnew = fmaxf(m[j], red_max16(fmaxf(x0, x1)));
            const float corr = __expf(m[j] - mnew);
            m[j] = mnew;
            const float p0 = __expf(x0 - mnew);
            const float p1 = __expf(x1 - mnew);
            l[j] = l[j] * corr + red_sum16(p0 + p1);
            acc0[j] *= corr; acc1[j] *= corr; acc2[j] *= corr; acc3[j] *= corr;

            const int prow = j + half8;
            pT[wave][prow * CHUNK + l15]      = (_Float16)p0;
            pT[wave][prow * CHUNK + 16 + l15] = (_Float16)p1;
        }

        // wave-private C-layout -> A-layout transpose via LDS (DS in-order per wave)
        asm volatile("s_wait_dscnt 0" ::: "memory");

        v16h pa;
        {
            const _Float16* pr = &pT[wave][l15 * CHUNK];
            v8h lo = *(const v8h*)(pr + half16);          // K 0-7  / 16-23
            v8h hh = *(const v8h*)(pr + half16 + 8);      // K 8-15 / 24-31
            #pragma unroll
            for (int e = 0; e < 8; ++e) { pa[e] = lo[e]; pa[8 + e] = hh[e]; }
        }

        // ---- O += P . V : B[k=t][n=d] from transposed V tile ----
        v16h vb0 = *(const v16h*)(&vT[(0 * 16 + l15) * CHUNK + half16]);
        v16h vb1 = *(const v16h*)(&vT[(1 * 16 + l15) * CHUNK + half16]);
        v16h vb2 = *(const v16h*)(&vT[(2 * 16 + l15) * CHUNK + half16]);
        v16h vb3 = *(const v16h*)(&vT[(3 * 16 + l15) * CHUNK + half16]);

        acc0 = __builtin_amdgcn_wmma_f32_16x16x32_f16(false, pa, false, vb0, (short)0, acc0, false, false);
        acc1 = __builtin_amdgcn_wmma_f32_16x16x32_f16(false, pa, false, vb1, (short)0, acc1, false, false);
        acc2 = __builtin_amdgcn_wmma_f32_16x16x32_f16(false, pa, false, vb2, (short)0, acc2, false, false);
        acc3 = __builtin_amdgcn_wmma_f32_16x16x32_f16(false, pa, false, vb3, (short)0, acc3, false, false);
    }

    // ---- epilogue: divide by row sum, store [B,S,H*D] f32 ----
    #pragma unroll
    for (int j = 0; j < 8; ++j) {
        const int grow = q0 + j + half8;
        const float rl = 1.0f / l[j];
        const size_t ob = (((size_t)b * S_ + grow) * H_ + h) * D_;
        out[ob + 0 * 16 + l15] = acc0[j] * rl;
        out[ob + 1 * 16 + l15] = acc1[j] * rl;
        out[ob + 2 * 16 + l15] = acc2[j] * rl;
        out[ob + 3 * 16 + l15] = acc3[j] * rl;
    }
}

extern "C" void kernel_launch(void* const* d_in, const int* in_sizes, int n_in,
                              void* d_out, int out_size, void* d_ws, size_t ws_size,
                              hipStream_t stream) {
    (void)in_sizes; (void)n_in; (void)out_size; (void)d_ws; (void)ws_size;
    const float*         q    = (const float*)d_in[0];
    const float*         k    = (const float*)d_in[1];
    const float*         v    = (const float*)d_in[2];
    const float*         bias = (const float*)d_in[3];
    const unsigned char* mask = (const unsigned char*)d_in[4];
    float* out = (float*)d_out;

    dim3 grid(S_ / (WAVES * 16), H_, B_);   // 32 x 16 x 2 = 1024 blocks
    dim3 block(WAVES * 32);                 // 4 waves of 32
    hipLaunchKernelGGL(core_attn_fwd, grid, block, 0, stream, q, k, v, bias, mask, out);
}
